// ConvolutionalAttention_1812476199063
// MI455X (gfx1250) — compile-verified, matrix operand order checked by reading, not run on hardware
//
#include <hip/hip_runtime.h>
#include <hip/hip_bf16.h>
#include <math.h>

// ---------------------------------------------------------------------------
// Problem constants (from reference)
// ---------------------------------------------------------------------------
#define BATCH 16
#define CTOT  64          // total channels in x / out
#define PDIM  32          // channels processed by the attention branch
#define HH    160
#define WW    160
#define HW    (HH * WW)   // 25600
#define LK    13
#define LPAD  6
#define SK    3
#define NTAP  (LK * LK)   // 169
#define HIDD  (PDIM / 2)  // 16
#define DKN   (PDIM * SK * SK) // 288

// Conv tiling: workgroup = 32 Cout x 8 rows x 32 px; 8 waves, one row each.
// Each wave owns 2 N-tiles x 2 M-tiles -> 4 WMMAs per tap per A-fragment pair.
#define OXT   32              // output x per workgroup (2 N-tiles of 16)
#define OYT   8               // output rows per workgroup (one per wave)
#define COLS  (OXT + LK - 1)  // 44 input cols staged
#define ROWSIN (OYT + LK - 1) // 20 input rows staged
#define FILL_ELEMS (ROWSIN * COLS * PDIM) // 28160 bf16 elems (= 256*110)

// Workspace layout (bytes)
#define WS_GAP_OFF   0                       // 512 floats
#define WS_DK_OFF    2048                    // 4608 floats
#define WS_APACK_OFF 20480                   // 169*2*32*16 bf16 = 346112 B

typedef __bf16 bf16_t;
typedef bf16_t v16bf __attribute__((ext_vector_type(16)));
typedef float  v8f   __attribute__((ext_vector_type(8)));

struct Q2 { uint4 a, b; };
static __device__ __forceinline__ v16bf make_frag(uint4 a, uint4 b) {
    Q2 p{a, b};
    return __builtin_bit_cast(v16bf, p);
}

static __device__ __forceinline__ unsigned short f2bf(float f) {
    unsigned int u = __builtin_bit_cast(unsigned int, f);
    unsigned int r = u + 0x7FFFu + ((u >> 16) & 1u);   // round-nearest-even
    return (unsigned short)(r >> 16);
}

// ---------------------------------------------------------------------------
// 1) Global average pool over H,W for x1 = x[:, :32]  -> gap (B, 32)
// ---------------------------------------------------------------------------
__global__ void gap_kernel(const float* __restrict__ x, float* __restrict__ gap) {
    __shared__ float red[256];
    const int blk = blockIdx.x;          // 0..511
    const int b = blk >> 5;              // /32
    const int c = blk & 31;
    const float* p = x + ((size_t)(b * CTOT + c)) * HW;
    float s = 0.f;
    for (int i = threadIdx.x; i < HW; i += 256) s += p[i];
    red[threadIdx.x] = s;
    __syncthreads();
    for (int off = 128; off > 0; off >>= 1) {
        if (threadIdx.x < off) red[threadIdx.x] += red[threadIdx.x + off];
        __syncthreads();
    }
    if (threadIdx.x == 0) gap[blk] = red[0] * (1.0f / (float)HW);
}

// ---------------------------------------------------------------------------
// 2) Tiny MLP: h = gelu(gap @ w1^T + b1); dk = h @ w2^T + b2   (one block)
// ---------------------------------------------------------------------------
__global__ void mlp_kernel(const float* __restrict__ gap,
                           const float* __restrict__ w1, const float* __restrict__ b1,
                           const float* __restrict__ w2, const float* __restrict__ b2,
                           float* __restrict__ dk) {
    __shared__ float sg[BATCH * PDIM];   // 512
    __shared__ float sh[BATCH * HIDD];   // 256
    for (int i = threadIdx.x; i < BATCH * PDIM; i += 256) sg[i] = gap[i];
    __syncthreads();
    {   // 16*16 = 256 hidden activations, one per thread
        const int b = threadIdx.x >> 4;
        const int j = threadIdx.x & 15;
        float acc = b1[j];
        #pragma unroll
        for (int ci = 0; ci < PDIM; ++ci) acc += sg[b * PDIM + ci] * w1[j * PDIM + ci];
        // exact gelu: 0.5*x*(1+erf(x/sqrt(2)))
        sh[threadIdx.x] = 0.5f * acc * (1.0f + erff(acc * 0.70710678118654752f));
    }
    __syncthreads();
    for (int o = threadIdx.x; o < BATCH * DKN; o += 256) {
        const int b = o / DKN;
        const int oo = o - b * DKN;
        float acc = b2[oo];
        #pragma unroll
        for (int j = 0; j < HIDD; ++j) acc += sh[b * HIDD + j] * w2[oo * HIDD + j];
        dk[o] = acc;
    }
}

// ---------------------------------------------------------------------------
// 3) Pre-pack lk_filter (32,32,13,13) fp32 -> bf16 in WMMA A-fragment order.
//    Layout: [tap t][mtile m][lane 0..31][half 0..15]
//    16-bit A layout (ISA 7.12.2):
//      lane<16 : M=lane,    halves 0..7 -> K=0..7,  halves 8..15 -> K=16..23
//      lane>=16: M=lane-16, halves 0..7 -> K=8..15, halves 8..15 -> K=24..31
//    K within a tap step == input channel ci.
// ---------------------------------------------------------------------------
__global__ void pack_filter_kernel(const float* __restrict__ lk,
                                   unsigned short* __restrict__ apack) {
    const int gid = blockIdx.x * 256 + threadIdx.x;   // 676*256 = 173056 exact
    if (gid >= NTAP * 2 * 32 * 16) return;
    const int h    = gid & 15;
    const int lane = (gid >> 4) & 31;
    const int m    = (gid >> 9) & 1;
    const int t    = gid >> 10;
    const int co = m * 16 + (lane & 15);
    const int k  = h + (h < 8 ? 0 : 8) + (lane < 16 ? 0 : 8); // = ci
    const int ky = t / LK, kx = t - ky * LK;
    const float v = lk[(((size_t)co * PDIM + k) * LK + ky) * LK + kx];
    apack[gid] = f2bf(v);
}

// ---------------------------------------------------------------------------
// 4) Main 13x13 dense conv as implicit GEMM with bf16 WMMA, f32 accumulate.
//    Grid: (160/OXT, 160/OYT, B) = (5, 20, 16).  Block: 256 threads = 8 waves.
//    Wave w computes output row r = w for all 32 Cout and all 32 px:
//      4 accumulators = [ntile 0/1] x [Cout 0..15 / 16..31].
//    Per tap: one A-fragment pair feeds 4 WMMAs (good A reuse from L2),
//    B fragments built with ds_load_b128 pairs from the bf16 LDS halo strip.
// ---------------------------------------------------------------------------
__global__ void __launch_bounds__(256)
conv_wmma_kernel(const float* __restrict__ x,
                 const uint4* __restrict__ apack,
                 float* __restrict__ out) {
    __shared__ uint4 smem[ROWSIN * COLS * PDIM * 2 / 16]; // 56320 B bf16 strip
    unsigned short* s16 = (unsigned short*)smem;

    const int b   = blockIdx.z;
    const int oy0 = blockIdx.y * OYT;
    const int ox0 = blockIdx.x * OXT;

    // ---- stage input halo strip (fp32 -> bf16), channel-contiguous ----
    // LDS element (ry, cx, ci) at index ((ry*COLS+cx)*PDIM + ci)
    for (int e = threadIdx.x; e < FILL_ELEMS; e += 256) {
        const int cx  = e % COLS;
        const int tmp = e / COLS;
        const int ry  = tmp % ROWSIN;
        const int ci  = tmp / ROWSIN;
        const int y  = oy0 + ry - LPAD;
        const int xg = ox0 + cx - LPAD;
        float v = 0.f;
        if (y >= 0 && y < HH && xg >= 0 && xg < WW)
            v = x[(((size_t)b * CTOT + ci) * HH + y) * WW + xg];
        s16[(ry * COLS + cx) * PDIM + ci] = f2bf(v);
    }
    __syncthreads();

    const int r    = threadIdx.x >> 5;   // wave index == output row in tile
    const int lane = threadIdx.x & 31;
    const int nx   = lane & 15;          // pixel within N-tile
    const int hi   = lane >> 4;          // selects ci bytes 0..15 / 16..31

    v8f acc00 = {};  // ntile 0, Cout  0..15
    v8f acc01 = {};  // ntile 0, Cout 16..31
    v8f acc10 = {};  // ntile 1, Cout  0..15
    v8f acc11 = {};  // ntile 1, Cout 16..31

    for (int ky = 0; ky < LK; ++ky) {
        const int rowbase = (r + ky) * COLS;
        #pragma unroll
        for (int kx = 0; kx < LK; ++kx) {
            const int t = ky * LK + kx;
            // A fragments (prepacked): per (t, m): 32 lanes x 2 uint4
            const uint4* ap = apack + (size_t)t * 128 + lane * 2;
            const v16bf a0 = make_frag(ap[0], ap[1]);
            const v16bf a1 = make_frag(ap[64], ap[65]);
            // B fragments from LDS for both N-tiles
            const int p0 = rowbase + nx + kx;
            const int i0 = p0 * 4 + hi;
            const v16bf bv0 = make_frag(smem[i0], smem[i0 + 2]);
            const int i1 = i0 + 64;   // +16 pixels * 4 uint4
            const v16bf bv1 = make_frag(smem[i1], smem[i1 + 2]);
            acc00 = __builtin_amdgcn_wmma_f32_16x16x32_bf16(false, a0, false, bv0,
                                                            (short)0, acc00, false, false);
            acc01 = __builtin_amdgcn_wmma_f32_16x16x32_bf16(false, a1, false, bv0,
                                                            (short)0, acc01, false, false);
            acc10 = __builtin_amdgcn_wmma_f32_16x16x32_bf16(false, a0, false, bv1,
                                                            (short)0, acc10, false, false);
            acc11 = __builtin_amdgcn_wmma_f32_16x16x32_bf16(false, a1, false, bv1,
                                                            (short)0, acc11, false, false);
        }
    }

    // ---- epilogue: D layout (ISA 7.12.2): VGPR v, lane l -> N=l%16, M=v+(l<16?0:8)
    const int oyg = oy0 + r;
    const int ox_n0 = ox0 + nx;
    const int ox_n1 = ox0 + 16 + nx;
    #pragma unroll
    for (int v = 0; v < 8; ++v) {
        const int co0 = v + hi * 8;
        float* o0 = out + (((size_t)b * CTOT + co0) * HH + oyg) * WW;
        float* o1 = out + (((size_t)b * CTOT + co0 + 16) * HH + oyg) * WW;
        o0[ox_n0] = acc00[v];
        o1[ox_n0] = acc01[v];
        o0[ox_n1] = acc10[v];
        o1[ox_n1] = acc11[v];
    }
}

// ---------------------------------------------------------------------------
// 5) fp32-exact dynamic depthwise 3x3 add (channels 0..31) + copy x2 (32..63)
// ---------------------------------------------------------------------------
__global__ void dyn_copy_kernel(const float* __restrict__ x,
                                const float* __restrict__ dk,
                                float* __restrict__ out) {
    const size_t i = (size_t)blockIdx.x * 256 + threadIdx.x;
    if (i >= (size_t)BATCH * CTOT * HW) return;
    const int xg = (int)(i % WW);
    const int y  = (int)((i / WW) % HH);
    const int c  = (int)((i / HW) % CTOT);
    const int b  = (int)(i / ((size_t)HW * CTOT));
    if (c >= PDIM) {
        out[i] = x[i];               // pass-through branch
        return;
    }
    const float* xp = x + (((size_t)b * CTOT + c) * HH) * WW;
    const float* k9 = dk + (size_t)b * DKN + c * (SK * SK);
    float s = 0.f;
    #pragma unroll
    for (int ky = 0; ky < SK; ++ky) {
        const int yy = y + ky - 1;
        if (yy < 0 || yy >= HH) continue;
        #pragma unroll
        for (int kx = 0; kx < SK; ++kx) {
            const int xx = xg + kx - 1;
            if (xx < 0 || xx >= WW) continue;
            s += xp[yy * WW + xx] * k9[ky * SK + kx];
        }
    }
    out[i] += s;                     // conv_wmma wrote x1_lk earlier this replay
}

// ---------------------------------------------------------------------------
extern "C" void kernel_launch(void* const* d_in, const int* in_sizes, int n_in,
                              void* d_out, int out_size, void* d_ws, size_t ws_size,
                              hipStream_t stream) {
    const float* x  = (const float*)d_in[0];
    const float* lk = (const float*)d_in[1];
    const float* w1 = (const float*)d_in[2];
    const float* b1 = (const float*)d_in[3];
    const float* w2 = (const float*)d_in[4];
    const float* b2 = (const float*)d_in[5];
    float* out = (float*)d_out;

    float* ws_gap = (float*)((char*)d_ws + WS_GAP_OFF);
    float* ws_dk  = (float*)((char*)d_ws + WS_DK_OFF);
    unsigned short* ws_apack = (unsigned short*)((char*)d_ws + WS_APACK_OFF);

    // 1) GAP: one block per (b, c)
    gap_kernel<<<BATCH * PDIM, 256, 0, stream>>>(x, ws_gap);
    // 2) MLP -> dynamic kernels
    mlp_kernel<<<1, 256, 0, stream>>>(ws_gap, w1, b1, w2, b2, ws_dk);
    // 3) Pack filter to WMMA fragment order (bf16)
    pack_filter_kernel<<<(NTAP * 2 * 32 * 16) / 256, 256, 0, stream>>>(lk, ws_apack);
    // 4) Main conv: implicit GEMM with v_wmma_f32_16x16x32_bf16
    dim3 cgrid(WW / OXT, HH / OYT, BATCH);
    conv_wmma_kernel<<<cgrid, 256, 0, stream>>>(x, (const uint4*)ws_apack, out);
    // 5) fp32 dynamic depthwise add + x2 copy
    const size_t tot = (size_t)BATCH * CTOT * HW;
    dyn_copy_kernel<<<(unsigned)((tot + 255) / 256), 256, 0, stream>>>(x, ws_dk, out);
}
